// MambaConvBlock_56324201120368
// MI455X (gfx1250) — matrix operand order repaired and checked by reading, not compiled, not run on hardware
//
#include <hip/hip_runtime.h>
#include <hip/hip_bf16.h>
#include <stdint.h>

// ---------------- CDNA5 / gfx1250 types -----------------------------------
typedef __attribute__((ext_vector_type(16))) __bf16 v16bf;
typedef __attribute__((ext_vector_type(8)))  float  v8f;
typedef __attribute__((ext_vector_type(4)))  unsigned v4u;
typedef __attribute__((ext_vector_type(8)))  int      v8i;
typedef __attribute__((ext_vector_type(4)))  int      v4i;

#define EPS_F 1e-5f

constexpr int Bn = 8, CIN = 64, COUT = 128, Hh = 64, Wd = 64;
constexpr int Lseq  = Hh * Wd;          // 4096
constexpr int Mtot  = Bn * Lseq;        // 32768 rows in all sequence GEMMs
constexpr int DSTATE = 16;
constexpr int DINNER = 256, DTRANK = 8;

// ---------------- Tensor Data Mover (TDM) staging --------------------------
#if __has_builtin(__builtin_amdgcn_tensor_load_to_lds)
#define HAVE_TDM 1
#else
#define HAVE_TDM 0
#endif

#if HAVE_TDM
// 2-D tile load: rows x rowElems bf16 elements, global row stride rowStride
// (elements), packed contiguously into LDS at byte offset ldsAddr.
// D# layout per CDNA5 ISA.
// This toolchain exposes the 6-arg builtin:
//   (uint32x4 g0, int32x8 g1, int32x4 g2, int32x4 g3, int32x8, i32 cpol)
__device__ __forceinline__ void tdm_load_2d(unsigned ldsAddr, const void* g,
                                            unsigned rowElems, unsigned rows,
                                            unsigned rowStride) {
  unsigned long long ga = (unsigned long long)g;
  v4u g0 = {1u,                                  // count=1, user mode
            ldsAddr,                             // lds_addr [63:32]
            (unsigned)ga,                        // global_addr [95:64]
            (unsigned)((ga >> 32) & 0x01FFFFFFu) // global_addr [120:96]
                | 0x80000000u};                  // type=2 ("image") [127:126]
  v8i g1;
  g1[0] = (int)(1u << 16);                                   // data_size=2B
  g1[1] = (int)((rowElems & 0xFFFFu) << 16);                 // tensor_dim0 lo
  g1[2] = (int)(((rowElems >> 16) & 0xFFFFu) |               // tensor_dim0 hi
                ((rows & 0xFFFFu) << 16));                   // tensor_dim1 lo
  g1[3] = (int)(((rows >> 16) & 0xFFFFu) |                   // tensor_dim1 hi
                ((rowElems & 0xFFFFu) << 16));               // tile_dim0
  g1[4] = (int)(rows & 0xFFFFu);                             // tile_dim1 (dim2=0)
  g1[5] = (int)rowStride;                                    // dim0_stride lo
  g1[6] = 0;                                                 // stride hi / dim1_stride
  g1[7] = 0;
  v4i z4 = {0, 0, 0, 0};
  v8i z8 = {0, 0, 0, 0, 0, 0, 0, 0};
  __builtin_amdgcn_tensor_load_to_lds(g0, g1, z4, z4, z8, 0);
}
#endif

// generic pointer to shared memory: LDS byte offset is addr[31:0] (ISA §10.2)
__device__ __forceinline__ unsigned lds_off(const void* p) {
  return (unsigned)(size_t)p;
}

// ---------------- fragment helpers (layouts per ISA 7.12.2) ----------------
__device__ __forceinline__ v16bf zero16bf() {
  union { uint4 u[2]; v16bf v; } t;
  t.u[0] = make_uint4(0u, 0u, 0u, 0u);
  t.u[1] = make_uint4(0u, 0u, 0u, 0u);
  return t.v;
}
// A-fragment: elements 0..7 <- p[0..7], elements 8..15 <- p[16..23]
__device__ __forceinline__ v16bf ld_split(const __bf16* p) {
  union { uint4 u[2]; v16bf v; } t;
  t.u[0] = *reinterpret_cast<const uint4*>(p);
  t.u[1] = *reinterpret_cast<const uint4*>(p + 16);
  return t.v;
}
// 16 consecutive halves (unused; B fragments use ld_split)
__device__ __forceinline__ v16bf ld_contig(const __bf16* p) {
  union { uint4 u[2]; v16bf v; } t;
  t.u[0] = *reinterpret_cast<const uint4*>(p);
  t.u[1] = *reinterpret_cast<const uint4*>(p + 8);
  return t.v;
}
__device__ __forceinline__ float silu_f(float x) { return x / (1.f + __expf(-x)); }

// ---------------- layout / weight prep kernels -----------------------------
__global__ __launch_bounds__(256) void nchw_to_nhwc_bf16(
    const float* __restrict__ x, __bf16* __restrict__ o) {
  int id = blockIdx.x * 256 + threadIdx.x;           // over B*H*W*CIN
  int c  = id & 63;
  int t  = id >> 6;
  int xw = t & 63;  t >>= 6;
  int y  = t & 63;
  int b  = t >> 6;
  o[id] = (__bf16)x[(((size_t)b * CIN + c) * Hh + y) * Wd + xw];
}

// conv weights (co,ci,3,3) f32 -> Wt[co][(kh*3+kw)*CI + ci] bf16
__global__ __launch_bounds__(256) void prep_conv_w(
    const float* __restrict__ w, __bf16* __restrict__ o, int CI) {
  int id = blockIdx.x * 256 + threadIdx.x;           // over 128*CI*9
  int co = id / (CI * 9);
  int t  = id % (CI * 9);
  int s  = t / CI;
  int ci = t % CI;
  o[id] = (__bf16)w[(size_t)(co * CI + ci) * 9 + s];
}

// dense weights [rows][cols] f32 -> [padrows][cols] bf16 (grid covers padrows*cols)
__global__ __launch_bounds__(256) void prep_mat_w(
    const float* __restrict__ w, __bf16* __restrict__ o, int rows, int cols) {
  int id = blockIdx.x * 256 + threadIdx.x;
  int r = id / cols, c = id % cols;
  o[id] = (r < rows) ? (__bf16)w[(size_t)r * cols + c] : (__bf16)0.0f;
}

__global__ __launch_bounds__(256) void neg_exp_k(
    const float* __restrict__ a, float* __restrict__ o, int n) {
  int id = blockIdx.x * 256 + threadIdx.x;
  if (id < n) o[id] = -__expf(a[id]);
}

// ---------------- implicit-GEMM conv3x3 + BN + ReLU (bf16 WMMA) ------------
// X: NHWC bf16 (CI), Wt: [co][9*CI] bf16, Y: (B*H*W, 128) bf16
template <int CI>
__global__ __launch_bounds__(256) void conv3x3_bn_relu(
    const __bf16* __restrict__ X, const __bf16* __restrict__ Wt,
    const float* __restrict__ cb, const float* __restrict__ g,
    const float* __restrict__ bet, const float* __restrict__ mu,
    const float* __restrict__ var, __bf16* __restrict__ Y) {
  constexpr int K9 = 9 * CI;
  __shared__ __bf16 sB[64 * CI];                 // one (kh,kw) slice of B tile

  const int tid  = threadIdx.x;
  const int wave = tid >> 5, lane = tid & 31;
  const int hf   = lane >> 4, l16 = lane & 15;
  const int row0 = blockIdx.x * 128 + wave * 16;
  const int ncol0 = blockIdx.y * 64;
#if HAVE_TDM
  const unsigned sBoff = lds_off(&sB[0]);
#endif

  // per-lane output pixel for the 16 A-rows of this wave's tile
  const int pid  = row0 + l16;
  const int pb   = pid >> 12;
  const int prem = pid & 4095;
  const int py   = prem >> 6, px = prem & 63;

  v8f acc[4];
#pragma unroll
  for (int f = 0; f < 4; ++f)
#pragma unroll
    for (int j = 0; j < 8; ++j) acc[f][j] = 0.f;

  for (int s = 0; s < 9; ++s) {
    __syncthreads();
#if HAVE_TDM
    // one TDM descriptor moves the whole 64 x CI slice into LDS
    if (tid < 32) {
      tdm_load_2d(sBoff, Wt + (size_t)ncol0 * K9 + s * CI, CI, 64, K9);
      __builtin_amdgcn_s_wait_tensorcnt(0);
    }
#else
    constexpr int CNT8 = 64 * CI / 8;
    for (int i = tid; i < CNT8; i += 256) {
      int col = i / (CI / 8);
      int k8  = (i % (CI / 8)) * 8;
      *reinterpret_cast<uint4*>(&sB[col * CI + k8]) =
          *reinterpret_cast<const uint4*>(
              &Wt[(size_t)(ncol0 + col) * K9 + s * CI + k8]);
    }
#endif
    __syncthreads();

    const int kh = s / 3, kw = s % 3;
    const int iy = py + kh - 1, ix = px + kw - 1;
    const bool valid = (iy >= 0) && (iy < Hh) && (ix >= 0) && (ix < Wd);
    const __bf16* ap =
        X + (size_t)((pb * Hh + iy) * Wd + ix) * CI + hf * 8;

#pragma unroll
    for (int kc = 0; kc < CI / 32; ++kc) {
      v16bf a = valid ? ld_split(ap + kc * 32) : zero16bf();
#pragma unroll
      for (int f = 0; f < 4; ++f) {
        v16bf b = ld_split(&sB[(f * 16 + l16) * CI + kc * 32 + hf * 8]);
        acc[f] = __builtin_amdgcn_wmma_f32_16x16x32_bf16(
            false, a, false, b, (short)0, acc[f], false, false);
      }
    }
  }

  // fused bias + BN(eval) + ReLU epilogue
#pragma unroll
  for (int f = 0; f < 4; ++f) {
    const int co = ncol0 + f * 16 + l16;
    const float sc = g[co] * rsqrtf(var[co] + EPS_F);
    const float sh = bet[co] + (cb[co] - mu[co]) * sc;
#pragma unroll
    for (int j = 0; j < 8; ++j) {
      const int m = row0 + hf * 8 + j;
      float v = acc[f][j] * sc + sh;
      v = v > 0.f ? v : 0.f;
      Y[(size_t)m * COUT + co] = (__bf16)v;
    }
  }
}

// ---------------- generic bf16 WMMA GEMM, weights staged in LDS -----------
// A: [Mtot][K] bf16 row-major, Bw: [N][K] bf16 ("n-major"), block = 8 waves
// MODE 0: C f32 -> outF[m*ldc + col] for col < Nvalid
// MODE 1: in_proj split: col<256 -> outB bf16 (xm); col>=256 -> outF = silu(z)
template <int K, int NT, int MODE>
__global__ __launch_bounds__(256) void gemm_bf16(
    const __bf16* __restrict__ A, const __bf16* __restrict__ Bw,
    float* __restrict__ outF, __bf16* __restrict__ outB,
    const int Nvalid, const int ldc) {
  constexpr int NC = NT * 16;
  __shared__ __bf16 sB[NC * K];

  const int tid  = threadIdx.x;
  const int wave = tid >> 5, lane = tid & 31;
  const int hf   = lane >> 4, l16 = lane & 15;
  const int row0 = blockIdx.x * 128 + wave * 16;
  const int ncol0 = blockIdx.y * NC;

#if HAVE_TDM
  // single TDM descriptor DMAs the whole NC x K bf16 weight tile to LDS
  if (tid < 32) {
    tdm_load_2d(lds_off(&sB[0]), Bw + (size_t)ncol0 * K, K, NC, K);
    __builtin_amdgcn_s_wait_tensorcnt(0);
  }
#else
  constexpr int CNT8 = NC * K / 8;
  for (int i = tid; i < CNT8; i += 256) {
    int col = i / (K / 8);
    int k8  = (i % (K / 8)) * 8;
    *reinterpret_cast<uint4*>(&sB[col * K + k8]) =
        *reinterpret_cast<const uint4*>(&Bw[(size_t)(ncol0 + col) * K + k8]);
  }
#endif
  __syncthreads();

  v8f acc[NT];
#pragma unroll
  for (int f = 0; f < NT; ++f)
#pragma unroll
    for (int j = 0; j < 8; ++j) acc[f][j] = 0.f;

  const __bf16* ap0 = A + (size_t)(row0 + l16) * K + hf * 8;
#pragma unroll
  for (int kc = 0; kc < K / 32; ++kc) {
    v16bf a = ld_split(ap0 + kc * 32);
#pragma unroll
    for (int f = 0; f < NT; ++f) {
      v16bf b = ld_split(&sB[(f * 16 + l16) * K + kc * 32 + hf * 8]);
      acc[f] = __builtin_amdgcn_wmma_f32_16x16x32_bf16(
          false, a, false, b, (short)0, acc[f], false, false);
    }
  }

#pragma unroll
  for (int f = 0; f < NT; ++f) {
    const int col = ncol0 + f * 16 + l16;
#pragma unroll
    for (int j = 0; j < 8; ++j) {
      const int m = row0 + hf * 8 + j;
      float v = acc[f][j];
      if (MODE == 0) {
        if (col < Nvalid) outF[(size_t)m * ldc + col] = v;
      } else {
        if (col < DINNER) outB[(size_t)m * DINNER + col] = (__bf16)v;
        else              outF[(size_t)m * DINNER + (col - DINNER)] = silu_f(v);
      }
    }
  }
}

// ---------------- depthwise causal conv1d + SiLU ---------------------------
__global__ __launch_bounds__(256) void dwconv_silu(
    const __bf16* __restrict__ xm, const float* __restrict__ w,
    const float* __restrict__ bias, float* __restrict__ of,
    __bf16* __restrict__ ob) {
  size_t id = (size_t)blockIdx.x * 256 + threadIdx.x;  // over Mtot*256
  int d = (int)(id & 255);
  size_t bl = id >> 8;
  int l = (int)(bl & 4095);
  int b = (int)(bl >> 12);
  float acc = bias[d];
  const float* wd = w + d * 4;
#pragma unroll
  for (int j = 0; j < 4; ++j) {
    int ls = l - 3 + j;
    if (ls >= 0)
      acc += wd[j] * (float)xm[(((size_t)b << 12) + ls) * DINNER + d];
  }
  float s = silu_f(acc);
  of[id] = s;
  ob[id] = (__bf16)s;
}

// ---------------- dt = softplus(dt_r @ dt_proj_w^T + b) --------------------
__global__ __launch_bounds__(256) void dt_softplus(
    const float* __restrict__ xdb, const float* __restrict__ dtw,
    const float* __restrict__ dtb, float* __restrict__ dt) {
  size_t id = (size_t)blockIdx.x * 256 + threadIdx.x;  // over Mtot*256
  int d = (int)(id & 255);
  size_t row = id >> 8;
  const float* r = xdb + row * 40;   // first DTRANK entries are dt_r
  float acc = dtb[d];
#pragma unroll
  for (int j = 0; j < DTRANK; ++j) acc += r[j] * dtw[d * DTRANK + j];
  dt[id] = fmaxf(acc, 0.f) + log1pf(__expf(-fabsf(acc)));
}

// ---------------- selective scan (sequential over L) -----------------------
// 1 block per batch, 1 lane per channel d; 16-state recurrence in registers.
// Fuses (+ D*x) * silu(z) and emits gated y in bf16 for out_proj.
__global__ __launch_bounds__(256) void ssm_scan(
    const float* __restrict__ dt, const float* __restrict__ xs,
    const float* __restrict__ xdb, const float* __restrict__ Aneg,
    const float* __restrict__ Dv, const float* __restrict__ sz,
    __bf16* __restrict__ yg) {
  const int b = blockIdx.x;
  const int d = threadIdx.x;
  float h[DSTATE], Ad[DSTATE];
#pragma unroll
  for (int s = 0; s < DSTATE; ++s) { h[s] = 0.f; Ad[s] = Aneg[d * DSTATE + s]; }
  const float Dd = Dv[d];

  for (int l = 0; l < Lseq; ++l) {
    const size_t r = (size_t)b * Lseq + l;
    const float* bc = xdb + r * 40;          // uniform across wave -> scalarizes
    __builtin_prefetch(bc + 40, 0, 1);       // next step's B/C row
    const float dt_t = dt[r * DINNER + d];
    const float x_t  = xs[r * DINNER + d];
    const float dx   = dt_t * x_t;
    float y = 0.f;
#pragma unroll
    for (int s = 0; s < DSTATE; ++s) {
      const float dA = __expf(dt_t * Ad[s]);
      h[s] = dA * h[s] + dx * bc[8 + s];     // B_t
      y = fmaf(h[s], bc[24 + s], y);         // C_t
    }
    yg[r * DINNER + d] = (__bf16)((y + Dd * x_t) * sz[r * DINNER + d]);
  }
}

// ---------------- LayerNorm over C=128 + transpose to NCHW -----------------
__global__ __launch_bounds__(256) void layernorm_out(
    const float* __restrict__ s, const float* __restrict__ g,
    const float* __restrict__ b, float* __restrict__ o) {
  const int wave = threadIdx.x >> 5, lane = threadIdx.x & 31;   // wave32
  const int row  = blockIdx.x * 8 + wave;                       // (b,l)
  const float* p = s + (size_t)row * COUT;
  float v[4], s1 = 0.f, s2 = 0.f;
#pragma unroll
  for (int t = 0; t < 4; ++t) {
    v[t] = p[lane + t * 32];
    s1 += v[t];
    s2 += v[t] * v[t];
  }
#pragma unroll
  for (int off = 16; off > 0; off >>= 1) {
    s1 += __shfl_xor(s1, off, 32);
    s2 += __shfl_xor(s2, off, 32);
  }
  const float mu  = s1 * (1.f / 128.f);
  const float var = s2 * (1.f / 128.f) - mu * mu;
  const float inv = rsqrtf(var + EPS_F);
  const int bb = row >> 12, l = row & 4095;
#pragma unroll
  for (int t = 0; t < 4; ++t) {
    int c = lane + t * 32;
    o[(((size_t)bb * COUT + c) << 12) + l] = (v[t] - mu) * inv * g[c] + b[c];
  }
}

// ---------------- host launcher --------------------------------------------
extern "C" void kernel_launch(void* const* d_in, const int* in_sizes, int n_in,
                              void* d_out, int out_size, void* d_ws,
                              size_t ws_size, hipStream_t stream) {
  (void)in_sizes; (void)n_in; (void)out_size; (void)ws_size;
  const float* x         = (const float*)d_in[0];
  const float* conv1_w   = (const float*)d_in[1];
  const float* conv1_b   = (const float*)d_in[2];
  const float* bn1_g     = (const float*)d_in[3];
  const float* bn1_b     = (const float*)d_in[4];
  const float* bn1_m     = (const float*)d_in[5];
  const float* bn1_v     = (const float*)d_in[6];
  const float* conv2_w   = (const float*)d_in[7];
  const float* conv2_b   = (const float*)d_in[8];
  const float* bn2_g     = (const float*)d_in[9];
  const float* bn2_b     = (const float*)d_in[10];
  const float* bn2_m     = (const float*)d_in[11];
  const float* bn2_v     = (const float*)d_in[12];
  const float* in_proj_w = (const float*)d_in[13];
  const float* conv1d_w  = (const float*)d_in[14];
  const float* conv1d_b  = (const float*)d_in[15];
  const float* x_proj_w  = (const float*)d_in[16];
  const float* dt_proj_w = (const float*)d_in[17];
  const float* dt_proj_b = (const float*)d_in[18];
  const float* A_log     = (const float*)d_in[19];
  const float* Dvec      = (const float*)d_in[20];
  const float* out_proj_w= (const float*)d_in[21];
  const float* ln_g      = (const float*)d_in[22];
  const float* ln_b      = (const float*)d_in[23];

  uint8_t* base = (uint8_t*)d_ws;
  size_t off = 0;
  auto alloc = [&](size_t bytes) -> void* {
    void* p = base + off;
    off += (bytes + 255) & ~(size_t)255;
    return p;
  };
  __bf16* xnhwc  = (__bf16*)alloc((size_t)Bn * Hh * Wd * CIN * 2);
  __bf16* wb1    = (__bf16*)alloc((size_t)COUT * 9 * CIN * 2);
  __bf16* wb2    = (__bf16*)alloc((size_t)COUT * 9 * COUT * 2);
  __bf16* wbin   = (__bf16*)alloc((size_t)512 * 128 * 2);
  __bf16* wbx    = (__bf16*)alloc((size_t)48 * 256 * 2);   // padded 40->48 rows
  __bf16* wbout  = (__bf16*)alloc((size_t)128 * 256 * 2);
  __bf16* h1     = (__bf16*)alloc((size_t)Mtot * 128 * 2);
  __bf16* seqb   = (__bf16*)alloc((size_t)Mtot * 128 * 2);
  __bf16* xmb    = (__bf16*)alloc((size_t)Mtot * 256 * 2);
  float*  szf    = (float*) alloc((size_t)Mtot * 256 * 4); // silu(z)
  float*  xsf    = (float*) alloc((size_t)Mtot * 256 * 4); // x_ssm f32
  __bf16* xsb    = (__bf16*)alloc((size_t)Mtot * 256 * 2); // x_ssm bf16
  float*  xdb    = (float*) alloc((size_t)Mtot * 40 * 4);
  float*  dtf    = (float*) alloc((size_t)Mtot * 256 * 4);
  float*  aneg   = (float*) alloc((size_t)256 * 16 * 4);
  __bf16* yg     = (__bf16*)alloc((size_t)Mtot * 256 * 2);
  float*  seqout = (float*) alloc((size_t)Mtot * 128 * 4);

  dim3 blk(256);

  nchw_to_nhwc_bf16<<<dim3((Bn * CIN * Hh * Wd) / 256), blk, 0, stream>>>(x, xnhwc);
  prep_conv_w<<<dim3((COUT * 9 * CIN) / 256), blk, 0, stream>>>(conv1_w, wb1, CIN);
  prep_conv_w<<<dim3((COUT * 9 * COUT) / 256), blk, 0, stream>>>(conv2_w, wb2, COUT);
  prep_mat_w<<<dim3((512 * 128) / 256), blk, 0, stream>>>(in_proj_w, wbin, 512, 128);
  prep_mat_w<<<dim3((48 * 256) / 256), blk, 0, stream>>>(x_proj_w, wbx, 40, 256);
  prep_mat_w<<<dim3((128 * 256) / 256), blk, 0, stream>>>(out_proj_w, wbout, 128, 256);
  neg_exp_k<<<dim3(16), blk, 0, stream>>>(A_log, aneg, 256 * 16);

  conv3x3_bn_relu<64><<<dim3(256, 2), blk, 0, stream>>>(
      xnhwc, wb1, conv1_b, bn1_g, bn1_b, bn1_m, bn1_v, h1);
  conv3x3_bn_relu<128><<<dim3(256, 2), blk, 0, stream>>>(
      h1, wb2, conv2_b, bn2_g, bn2_b, bn2_m, bn2_v, seqb);

  // in_proj: (32768,128)x(128,512); split xm bf16 / silu(z) f32
  gemm_bf16<128, 4, 1><<<dim3(256, 8), blk, 0, stream>>>(
      seqb, wbin, szf, xmb, 512, 256);

  dwconv_silu<<<dim3((Mtot * 256) / 256), blk, 0, stream>>>(
      xmb, conv1d_w, conv1d_b, xsf, xsb);

  // x_proj: (32768,256)x(256,40) padded to 48 cols
  gemm_bf16<256, 3, 0><<<dim3(256, 1), blk, 0, stream>>>(
      xsb, wbx, xdb, nullptr, 40, 40);

  dt_softplus<<<dim3((Mtot * 256) / 256), blk, 0, stream>>>(
      xdb, dt_proj_w, dt_proj_b, dtf);

  ssm_scan<<<dim3(8), blk, 0, stream>>>(dtf, xsf, xdb, aneg, Dvec, szf, yg);

  // out_proj: (32768,256)x(256,128)
  gemm_bf16<256, 4, 0><<<dim3(256, 2), blk, 0, stream>>>(
      yg, wbout, seqout, nullptr, 128, 128);

  layernorm_out<<<dim3(Mtot / 8), blk, 0, stream>>>(
      seqout, ln_g, ln_b, (float*)d_out);
}
